// GRCU_RGCN_87909390614844
// MI455X (gfx1250) — matrix operand order repaired and checked by reading, not compiled, not run on hardware
//
#include <hip/hip_runtime.h>
#include <math.h>

// Problem constants (match reference)
#define TT  3
#define NN  20000
#define FF  256
#define FOO 256
#define EE  320000
#define RRL 4

typedef float  v2f   __attribute__((ext_vector_type(2)));
typedef float  v8f   __attribute__((ext_vector_type(8)));
typedef __bf16 v16bf __attribute__((ext_vector_type(16)));
typedef unsigned int u32x4 __attribute__((ext_vector_type(4)));

union BV16 { v16bf v; u32x4 q[2]; };

// ---------------------------------------------------------------------------
// gfx1250 async global->LDS copy helpers (inline asm: portable across
// ROCm 7.2 and amdgpu-toolchain, bypasses the differing builtin arity).
// ---------------------------------------------------------------------------
__device__ __forceinline__ void grcu_async_ld16(unsigned lds_off, const void* gptr) {
    asm volatile("global_load_async_to_lds_b128 %0, %1, off"
                 :: "v"(lds_off), "v"(gptr)
                 : "memory");
}
__device__ __forceinline__ void grcu_wait_async0() {
    asm volatile("s_wait_asynccnt 0x0" ::: "memory");
}

// ---------------------------------------------------------------------------
// Small utility kernels
// ---------------------------------------------------------------------------
__global__ void grcu_copy_f32(const float* __restrict__ s, float* __restrict__ d, int n) {
    int i = blockIdx.x * blockDim.x + threadIdx.x;
    if (i < n) d[i] = s[i];
}

__global__ void grcu_zero_f32(float* __restrict__ d, int n) {
    int i = blockIdx.x * blockDim.x + threadIdx.x;
    if (i < n) d[i] = 0.0f;
}

__global__ void grcu_cvt_bf16(const float* __restrict__ s, __bf16* __restrict__ d, int n) {
    int i = blockIdx.x * blockDim.x + threadIdx.x;
    if (i < n) d[i] = (__bf16)s[i];
}

// 1/(||scorer||+1e-8), scorer has 256 elements
__global__ void grcu_snorm(const float* __restrict__ scorer, float* __restrict__ inv) {
    __shared__ float sh[256];
    float v = scorer[threadIdx.x];
    sh[threadIdx.x] = v * v;
    __syncthreads();
    for (int off = 128; off > 0; off >>= 1) {
        if (threadIdx.x < (unsigned)off) sh[threadIdx.x] += sh[threadIdx.x + off];
        __syncthreads();
    }
    if (threadIdx.x == 0) inv[0] = 1.0f / (sqrtf(sh[0]) + 1e-8f);
}

// scores[n] = dot(xt[n,:], scorer)*inv + mask[n].  One wave32 per node.
__global__ __launch_bounds__(128)
void grcu_scores(const float* __restrict__ xt, const float* __restrict__ scorer,
                 const float* __restrict__ inv, const float* __restrict__ mask_t,
                 float* __restrict__ scores, int n) {
    int lane = threadIdx.x & 31;
    int w    = threadIdx.x >> 5;
    int node = blockIdx.x * 4 + w;
    if (node >= n) return;
    const float* xr = xt + (size_t)node * FF;
    float s = 0.0f;
#pragma unroll
    for (int i = 0; i < 8; ++i) s += xr[lane + 32 * i] * scorer[lane + 32 * i];
#pragma unroll
    for (int off = 16; off > 0; off >>= 1) s += __shfl_down(s, off, 32);
    if (lane == 0) scores[node] = s * inv[0] + mask_t[node];
}

// ---------------------------------------------------------------------------
// Tournament top-256: each block bitonic-sorts a 1024-element chunk of
// (value, index) pairs (value desc, index asc for ties -> lax.top_k order)
// and emits its top 256.  Repeated 4 levels: 20000 -> 5120 -> 1280 -> 512 -> 256.
// iin == nullptr means "index = global position".
// ---------------------------------------------------------------------------
__global__ __launch_bounds__(1024)
void grcu_sort_top256(const float* __restrict__ vin, const int* __restrict__ iin,
                      int n_in, float* __restrict__ vout, int* __restrict__ iout) {
    __shared__ float sv[1024];
    __shared__ int   si[1024];
    const int tid  = threadIdx.x;
    const int gpos = blockIdx.x * 1024 + tid;
    float v = -INFINITY;
    int   ix = 0x7fffffff;
    if (gpos < n_in) { v = vin[gpos]; ix = iin ? iin[gpos] : gpos; }
    sv[tid] = v; si[tid] = ix;
    __syncthreads();
    for (int k = 2; k <= 1024; k <<= 1) {
        for (int j = k >> 1; j > 0; j >>= 1) {
            int p = tid ^ j;
            if (p > tid) {
                float v0 = sv[tid], v1 = sv[p];
                int   i0 = si[tid], i1 = si[p];
                bool before = (v0 > v1) || (v0 == v1 && i0 < i1);
                bool up = (tid & k) == 0;
                if (up ? !before : before) {
                    sv[tid] = v1; sv[p] = v0;
                    si[tid] = i1; si[p] = i0;
                }
            }
            __syncthreads();
        }
    }
    if (tid < 256) {
        vout[blockIdx.x * 256 + tid] = sv[tid];
        iout[blockIdx.x * 256 + tid] = si[tid];
    }
}

__global__ void grcu_topk_fin(const float* __restrict__ v, const int* __restrict__ ix,
                              int* __restrict__ tidx, float* __restrict__ ttanh) {
    int j = threadIdx.x;
    tidx[j]  = ix[j];
    ttanh[j] = tanhf(v[j]);
}

// zt[f, j] = xt[idx[j], f] * tanh(vals[j]); zt row-major [FF, FOO]
__global__ void grcu_build_zt(const float* __restrict__ xt, const int* __restrict__ idx,
                              const float* __restrict__ tanhv, float* __restrict__ zt) {
    int j = blockIdx.x;       // 0..255 column (rank)
    int f = threadIdx.x;      // 0..255 feature
    zt[f * FOO + j] = xt[(size_t)idx[j] * FF + f] * tanhv[j];
}

// ---------------------------------------------------------------------------
// 256x256x256 f32 GEMM via V_WMMA_F32_16X16X4_F32.
// Block = 128 threads = 4 waves, each wave a 16x16 tile (32x32 per block).
// Plain variant: D = A@B.  Fused variant: D = A@B + add0 + add1 (no branches).
// ---------------------------------------------------------------------------
__device__ __forceinline__ v8f grcu_gemm256_core(const float* __restrict__ A,
                                                 const float* __restrict__ B,
                                                 int m0, int n0, int g, int ml) {
    v8f c = {0.f,0.f,0.f,0.f,0.f,0.f,0.f,0.f};
    const float* arow = A + (size_t)(m0 + ml) * 256;
    for (int k = 0; k < 256; k += 4) {
        v2f a, b;
        // A 16x4 layout: lane m, group g -> {A[m][k+2g], A[m][k+2g+1]}
        a.x = arow[k + 2 * g];
        a.y = arow[k + 2 * g + 1];
        // B 4x16 layout: VGPR v, group g -> B[k+2g+v][n]
        b.x = B[(size_t)(k + 2 * g) * 256 + n0 + ml];
        b.y = B[(size_t)(k + 2 * g + 1) * 256 + n0 + ml];
        c = __builtin_amdgcn_wmma_f32_16x16x4_f32(false, a, false, b, (short)0, c,
                                                  false, false);
    }
    return c;
}

__global__ __launch_bounds__(128)
void grcu_gemm256_f32(const float* __restrict__ A, const float* __restrict__ B,
                      float* __restrict__ D) {
    const int lane = threadIdx.x & 31;
    const int w    = threadIdx.x >> 5;
    const int m0   = blockIdx.y * 32 + (w >> 1) * 16;
    const int n0   = blockIdx.x * 32 + (w & 1) * 16;
    const int g    = lane >> 4;
    const int ml   = lane & 15;
    v8f c = grcu_gemm256_core(A, B, m0, n0, g, ml);
#pragma unroll
    for (int v = 0; v < 8; ++v) {
        int row = m0 + v + 8 * g;                       // C/D: VGPR v -> row v+8g
        D[(size_t)row * 256 + n0 + ml] = c[v];
    }
}

__global__ __launch_bounds__(128)
void grcu_gemm256_fused(const float* __restrict__ A, const float* __restrict__ B,
                        const float* __restrict__ add0, const float* __restrict__ add1,
                        float* __restrict__ D) {
    const int lane = threadIdx.x & 31;
    const int w    = threadIdx.x >> 5;
    const int m0   = blockIdx.y * 32 + (w >> 1) * 16;
    const int n0   = blockIdx.x * 32 + (w & 1) * 16;
    const int g    = lane >> 4;
    const int ml   = lane & 15;
    v8f c = grcu_gemm256_core(A, B, m0, n0, g, ml);
#pragma unroll
    for (int v = 0; v < 8; ++v) {
        int row = m0 + v + 8 * g;
        size_t o = (size_t)row * 256 + n0 + ml;
        D[o] = c[v] + add0[o] + add1[o];
    }
}

// ---------------------------------------------------------------------------
// Pack a 256x256 f32 matrix (row major, [K,N]) into bf16 WMMA-B layout:
// Bpack[((kt*16+nt)*32+lane)*16 + e] = (bf16)Q[kt*32 + 16*(lane/16) + e][nt*16 + lane%16]
// ---------------------------------------------------------------------------
__global__ void grcu_pack_b_bf16(const float* __restrict__ Q, __bf16* __restrict__ Bp) {
    int tid  = blockIdx.x * blockDim.x + threadIdx.x;   // 0..65535
    int kt   = tid >> 13;
    int rem  = tid & 8191;
    int nt   = rem >> 9;
    int rem2 = rem & 511;
    int lane = rem2 >> 4;
    int e    = rem2 & 15;
    int K    = kt * 32 + (lane >> 4) * 16 + e;
    int Nc   = nt * 16 + (lane & 15);
    Bp[tid] = (__bf16)Q[(size_t)K * 256 + Nc];
}

// ---------------------------------------------------------------------------
// Big GEMM: D[N,256] = Abf[N,256] @ Q (Q pre-packed bf16) via
// V_WMMA_F32_16X16X32_BF16.  Block = 256 threads = 8 waves; block computes a
// 16-row x 256-col slab.  The 8 KB A-slab is staged once into LDS with
// gfx1250 async global->LDS b128 copies, then waves read WMMA A-fragments
// with ds_load_b128 (no 8x redundant global traffic).
// ---------------------------------------------------------------------------
__global__ __launch_bounds__(256)
void grcu_gemm_big_bf16(const __bf16* __restrict__ Abf, const __bf16* __restrict__ Bp,
                        float* __restrict__ D) {
    __shared__ __align__(16) __bf16 Asl[16 * 256];      // 8 KB slab
    const int tid  = threadIdx.x;
    const int lane = tid & 31;
    const int w    = tid >> 5;                          // 0..7
    const int m0   = blockIdx.x * 16;
    const int n0   = w * 32;
    const int g    = lane >> 4;
    const int ml   = lane & 15;

    // Async stage: 16 rows x 512 B, contiguous in global.  256 threads x 2 x 16 B.
    {
        const char* gA = (const char*)(Abf + (size_t)m0 * 256);
        unsigned lds0 = (unsigned)(size_t)(&Asl[0]);
        grcu_async_ld16(lds0 + (unsigned)tid * 16u,         gA + (size_t)tid * 16u);
        grcu_async_ld16(lds0 + 4096u + (unsigned)tid * 16u, gA + 4096u + (size_t)tid * 16u);
        grcu_wait_async0();
        __syncthreads();
    }

    v8f c0 = {0.f,0.f,0.f,0.f,0.f,0.f,0.f,0.f};
    v8f c1 = {0.f,0.f,0.f,0.f,0.f,0.f,0.f,0.f};
    const __bf16* arow = Asl + ml * 256;

#pragma unroll
    for (int kt = 0; kt < 8; ++kt) {
        const int k0 = kt * 32;
        BV16 a, b0, b1;
        // A bf16 16x32: elems 0..15 -> K=k0+16g+e
        a.q[0] = *(const u32x4*)(arow + k0 + 16 * g);
        a.q[1] = *(const u32x4*)(arow + k0 + 16 * g + 8);
        // B pre-packed: 32 contiguous bytes per lane per (kt, nt) tile
        const __bf16* bp0 = Bp + (((size_t)(kt * 16 + (n0 >> 4)) * 32 + lane) << 4);
        b0.q[0] = *(const u32x4*)(bp0);
        b0.q[1] = *(const u32x4*)(bp0 + 8);
        const __bf16* bp1 = bp0 + 512;                   // next 16-col tile
        b1.q[0] = *(const u32x4*)(bp1);
        b1.q[1] = *(const u32x4*)(bp1 + 8);
        c0 = __builtin_amdgcn_wmma_f32_16x16x32_bf16(false, a.v, false, b0.v, (short)0,
                                                     c0, false, false);
        c1 = __builtin_amdgcn_wmma_f32_16x16x32_bf16(false, a.v, false, b1.v, (short)0,
                                                     c1, false, false);
    }
#pragma unroll
    for (int v = 0; v < 8; ++v) {
        int row = m0 + v + 8 * g;
        D[(size_t)row * 256 + n0 + ml]      = c0[v];
        D[(size_t)row * 256 + n0 + 16 + ml] = c1[v];
    }
}

// ---------------------------------------------------------------------------
// GRU gate elementwise pieces
// ---------------------------------------------------------------------------
__global__ void grcu_gates(float* __restrict__ ZP, float* __restrict__ RP,
                           const float* __restrict__ Q, float* __restrict__ RQ, int n) {
    int i = blockIdx.x * blockDim.x + threadIdx.x;
    if (i < n) {
        float u = 1.0f / (1.0f + expf(-ZP[i]));
        float r = 1.0f / (1.0f + expf(-RP[i]));
        ZP[i] = u;
        RP[i] = r;
        RQ[i] = r * Q[i];
    }
}

__global__ void grcu_final(float* __restrict__ Q, const float* __restrict__ upd,
                           const float* __restrict__ HP, int n) {
    int i = blockIdx.x * blockDim.x + threadIdx.x;
    if (i < n) {
        float u = upd[i];
        float h = tanhf(HP[i]);
        Q[i] = (1.0f - u) * Q[i] + u * h;
    }
}

// ---------------------------------------------------------------------------
// Edge machinery
// ---------------------------------------------------------------------------
__global__ void grcu_deg(const int* __restrict__ row, const int* __restrict__ et,
                         float* __restrict__ deg4, int e_cnt, int n) {
    int e = blockIdx.x * blockDim.x + threadIdx.x;
    if (e < e_cnt) atomicAdd(&deg4[(size_t)et[e] * n + row[e]], 1.0f);
}

__global__ void grcu_dis(float* __restrict__ deg4, int n) {
    int i = blockIdx.x * blockDim.x + threadIdx.x;
    if (i < n) {
        float d = deg4[i];
        deg4[i] = (d > 0.0f) ? rsqrtf(d) : 0.0f;
    }
}

// 64 threads per edge (4 floats each): out[row] += Y[col] * norm
__global__ void grcu_edge_msg(const int* __restrict__ row, const int* __restrict__ col,
                              const int* __restrict__ et, const float* __restrict__ dis,
                              const float* __restrict__ Y, float* __restrict__ outt,
                              int e_cnt, int r) {
    long long tid = (long long)blockIdx.x * blockDim.x + threadIdx.x;
    int e = (int)(tid >> 6);
    if (e >= e_cnt) return;
    if (et[e] != r) return;
    int c4 = ((int)tid & 63) << 2;
    int ro = row[e], co = col[e];
    float nrm = dis[ro] * dis[co];
    const float* y = Y + (size_t)co * 256 + c4;
    float* o = outt + (size_t)ro * 256 + c4;
    atomicAdd(&o[0], y[0] * nrm);
    atomicAdd(&o[1], y[1] * nrm);
    atomicAdd(&o[2], y[2] * nrm);
    atomicAdd(&o[3], y[3] * nrm);
}

__global__ void grcu_relu(float* __restrict__ d, int n) {
    int i = blockIdx.x * blockDim.x + threadIdx.x;
    if (i < n) d[i] = fmaxf(d[i], 0.0f);
}

// ---------------------------------------------------------------------------
// Host orchestration
// ---------------------------------------------------------------------------
extern "C" void kernel_launch(void* const* d_in, const int* in_sizes, int n_in,
                              void* d_out, int out_size, void* d_ws, size_t ws_size,
                              hipStream_t stream) {
    const float* x           = (const float*)d_in[0];
    const float* mask        = (const float*)d_in[1];
    const float* W_self_init = (const float*)d_in[2];
    const float* W_rel_init  = (const float*)d_in[3];
    const float* sWz = (const float*)d_in[4];
    const float* sUz = (const float*)d_in[5];
    const float* sbz = (const float*)d_in[6];
    const float* sWr = (const float*)d_in[7];
    const float* sUr = (const float*)d_in[8];
    const float* sbr = (const float*)d_in[9];
    const float* sWh = (const float*)d_in[10];
    const float* sUh = (const float*)d_in[11];
    const float* sbh = (const float*)d_in[12];
    const float* s_scorer = (const float*)d_in[13];
    const float* rWz = (const float*)d_in[14];
    const float* rUz = (const float*)d_in[15];
    const float* rbz = (const float*)d_in[16];
    const float* rWr = (const float*)d_in[17];
    const float* rUr = (const float*)d_in[18];
    const float* rbr = (const float*)d_in[19];
    const float* rWh = (const float*)d_in[20];
    const float* rUh = (const float*)d_in[21];
    const float* rbh = (const float*)d_in[22];
    const float* r_scorer = (const float*)d_in[23];
    const int* edge_index = (const int*)d_in[24];
    const int* edge_type  = (const int*)d_in[25];

    // ---- workspace carve ----
    char* base = (char*)d_ws;
    size_t off = 0;
    auto carve = [&](size_t bytes) -> void* {
        void* p = base + off;
        off = (off + bytes + 255) & ~(size_t)255;
        return p;
    };
    const int M = FF * FOO;                     // 65536
    float* Q      = (float*)carve((size_t)5 * M * 4);   // Qs, Qr[0..3]
    float* zt     = (float*)carve((size_t)M * 4);
    float* G1     = (float*)carve((size_t)M * 4);
    float* G2     = (float*)carve((size_t)M * 4);
    float* G3     = (float*)carve((size_t)M * 4);
    float* ZP     = (float*)carve((size_t)M * 4);
    float* RP     = (float*)carve((size_t)M * 4);
    float* HP     = (float*)carve((size_t)M * 4);
    float* RQ     = (float*)carve((size_t)M * 4);
    float* scores = (float*)carve((size_t)NN * 4);
    float* candVA = (float*)carve(5120 * 4);
    int*   candIA = (int*)carve(5120 * 4);
    float* candVB = (float*)carve(5120 * 4);
    int*   candIB = (int*)carve(5120 * 4);
    int*   tidx   = (int*)carve(256 * 4);
    float* ttanh  = (float*)carve(256 * 4);
    float* sinv   = (float*)carve(4);
    __bf16* xbf   = (__bf16*)carve((size_t)NN * FF * 2);
    __bf16* bpack = (__bf16*)carve((size_t)M * 2);
    float* Y      = (float*)carve((size_t)NN * FOO * 4);
    float* deg4   = (float*)carve((size_t)RRL * NN * 4);
    (void)ws_size; (void)n_in; (void)in_sizes; (void)out_size;

    // per-cell weight pointers (cell 0 = self, cells 1..4 = relations)
    const float *Wz[5], *Uz[5], *bz[5], *Wr[5], *Ur[5], *br[5], *Wh[5], *Uh[5], *bh[5], *sc[5];
    Wz[0]=sWz; Uz[0]=sUz; bz[0]=sbz; Wr[0]=sWr; Ur[0]=sUr; br[0]=sbr;
    Wh[0]=sWh; Uh[0]=sUh; bh[0]=sbh; sc[0]=s_scorer;
    for (int r = 0; r < RRL; ++r) {
        Wz[r+1]=rWz+(size_t)r*M; Uz[r+1]=rUz+(size_t)r*M; bz[r+1]=rbz+(size_t)r*M;
        Wr[r+1]=rWr+(size_t)r*M; Ur[r+1]=rUr+(size_t)r*M; br[r+1]=rbr+(size_t)r*M;
        Wh[r+1]=rWh+(size_t)r*M; Uh[r+1]=rUh+(size_t)r*M; bh[r+1]=rbh+(size_t)r*M;
        sc[r+1]=r_scorer+(size_t)r*FF;
    }

    // ---- init Q state ----
    grcu_copy_f32<<<(M + 255) / 256, 256, 0, stream>>>(W_self_init, Q, M);
    grcu_copy_f32<<<(4 * M + 255) / 256, 256, 0, stream>>>(W_rel_init, Q + M, 4 * M);

    const dim3 g8(8, 8);
    for (int t = 0; t < TT; ++t) {
        const float* xt     = x + (size_t)t * NN * FF;
        const float* mask_t = mask + (size_t)t * NN;
        const int* row = edge_index + (size_t)t * 2 * EE;
        const int* col = row + EE;
        const int* et  = edge_type + (size_t)t * EE;
        float* outt = (float*)d_out + (size_t)t * NN * FOO;

        grcu_cvt_bf16<<<(NN * FF) / 256, 256, 0, stream>>>(xt, xbf, NN * FF);

        // ---- 5 GRU cells ----
        for (int c = 0; c < 5; ++c) {
            float* Qc = Q + (size_t)c * M;
            grcu_snorm<<<1, 256, 0, stream>>>(sc[c], sinv);
            grcu_scores<<<NN / 4, 128, 0, stream>>>(xt, sc[c], sinv, mask_t, scores, NN);
            // tournament top-256: 20000 -> 5120 -> 1280 -> 512 -> 256 (sorted)
            grcu_sort_top256<<<20, 1024, 0, stream>>>(scores, nullptr, NN, candVA, candIA);
            grcu_sort_top256<<<5, 1024, 0, stream>>>(candVA, candIA, 5120, candVB, candIB);
            grcu_sort_top256<<<2, 1024, 0, stream>>>(candVB, candIB, 1280, candVA, candIA);
            grcu_sort_top256<<<1, 1024, 0, stream>>>(candVA, candIA, 512, candVB, candIB);
            grcu_topk_fin<<<1, 256, 0, stream>>>(candVB, candIB, tidx, ttanh);
            grcu_build_zt<<<256, 256, 0, stream>>>(xt, tidx, ttanh, zt);

            grcu_gemm256_f32<<<g8, 128, 0, stream>>>(Wz[c], zt, G1);
            grcu_gemm256_f32<<<g8, 128, 0, stream>>>(Wr[c], zt, G2);
            grcu_gemm256_f32<<<g8, 128, 0, stream>>>(Wh[c], zt, G3);
            grcu_gemm256_fused<<<g8, 128, 0, stream>>>(Uz[c], Qc, G1, bz[c], ZP);
            grcu_gemm256_fused<<<g8, 128, 0, stream>>>(Ur[c], Qc, G2, br[c], RP);
            grcu_gates<<<M / 256, 256, 0, stream>>>(ZP, RP, Qc, RQ, M);
            grcu_gemm256_fused<<<g8, 128, 0, stream>>>(Uh[c], RQ, G3, bh[c], HP);
            grcu_final<<<M / 256, 256, 0, stream>>>(Qc, ZP, HP, M);
        }

        // ---- out = xt @ Qs (written straight into d_out slice) ----
        grcu_pack_b_bf16<<<M / 256, 256, 0, stream>>>(Q, bpack);
        grcu_gemm_big_bf16<<<NN / 16, 256, 0, stream>>>(xbf, bpack, outt);

        // ---- degree / normalization ----
        grcu_zero_f32<<<(RRL * NN + 255) / 256, 256, 0, stream>>>(deg4, RRL * NN);
        grcu_deg<<<(EE + 255) / 256, 256, 0, stream>>>(row, et, deg4, EE, NN);
        grcu_dis<<<(RRL * NN + 255) / 256, 256, 0, stream>>>(deg4, RRL * NN);

        // ---- relation messages ----
        for (int r = 0; r < RRL; ++r) {
            grcu_pack_b_bf16<<<M / 256, 256, 0, stream>>>(Q + (size_t)(r + 1) * M, bpack);
            grcu_gemm_big_bf16<<<NN / 16, 256, 0, stream>>>(xbf, bpack, Y);
            grcu_edge_msg<<<(EE * 64) / 256, 256, 0, stream>>>(row, col, et,
                                                              deg4 + (size_t)r * NN,
                                                              Y, outt, EE, r);
        }

        grcu_relu<<<(NN * FOO) / 256, 256, 0, stream>>>(outt, NN * FOO);
    }
}
